// MoE_32332513804634
// MI455X (gfx1250) — compile-verified, matrix operand order checked by reading, not run on hardware
//
#include <hip/hip_runtime.h>

// ---------------- problem constants ----------------
#define T_TOK 4096          // 4 * 1024 tokens
#define DIM   2048          // embedding dim D
#define HID   1408          // moe_inter_dim H
#define HSH   2816          // shared hidden HS = 2*1408
#define NEXP  16            // routed experts
#define TOPK  6

// ---------------- GEMM tiling ----------------
#define BM 128              // block M tile (tokens)
#define BN 128              // block N tile (output features)
#define BK 64               // K chunk staged in LDS per step
#define LDS_ROW 72          // u16 per LDS row: 64 data + 8 pad (bank spread)

typedef unsigned short u16;
typedef unsigned int   u32;

typedef __attribute__((ext_vector_type(16))) __bf16 v16bf;
typedef __attribute__((ext_vector_type(2)))  __bf16 v2bf;
typedef __attribute__((ext_vector_type(8)))  float  v8f;

union FragBF {
    v16bf bf;
    u16   u[16];
    u32   w32[8];
    uint4 q[2];
};

// f32 pair -> packed bf16x2 (single v_cvt_pk_bf16_f32)
static __device__ __forceinline__ u32 pk2bf(float lo, float hi) {
#if __has_builtin(__builtin_amdgcn_cvt_pk_bf16_f32)
    auto v = __builtin_amdgcn_cvt_pk_bf16_f32(lo, hi);
    return __builtin_bit_cast(u32, v);
#else
    v2bf v = { (__bf16)lo, (__bf16)hi };
    return __builtin_bit_cast(u32, v);
#endif
}
static __device__ __forceinline__ u16 f2bf(float f) {
    __bf16 b = (__bf16)f;
    return __builtin_bit_cast(u16, b);
}

// Fragment load from an LDS row: elements 0..7 at p, 8..15 at p+16
// (CDNA5 16-bit A/B VGPR layout; caller offsets p by half*8).
static __device__ __forceinline__ void load_frag_lds(FragBF& f, const u16* p) {
    f.q[0] = *(const uint4*)(p);
    f.q[1] = *(const uint4*)(p + 16);
}

// Stage 32 f32 -> 32 bf16 into LDS (4x ds_store_b128), with prefetch of the
// next K-chunk of this row.
static __device__ __forceinline__ void stage_row_f32(u16* dst, const float* __restrict__ src) {
    __builtin_prefetch(src + BK, 0, 3);
    uint4* d = (uint4*)dst;
#pragma unroll
    for (int i = 0; i < 4; ++i) {
        const float4 p0 = *(const float4*)(src + i * 8);
        const float4 p1 = *(const float4*)(src + i * 8 + 4);
        uint4 o;
        o.x = pk2bf(p0.x, p0.y); o.y = pk2bf(p0.z, p0.w);
        o.z = pk2bf(p1.x, p1.y); o.w = pk2bf(p1.z, p1.w);
        d[i] = o;
    }
}
// Stage 32 bf16 (already converted) into LDS (4x ds_store_b128).
static __device__ __forceinline__ void stage_row_bf16(u16* dst, const u16* __restrict__ src) {
    uint4* d = (uint4*)dst;
    const uint4* s = (const uint4*)src;
#pragma unroll
    for (int i = 0; i < 4; ++i) d[i] = s[i];
}

#define WMMA_BF16(A, B, C) \
    __builtin_amdgcn_wmma_f32_16x16x32_bf16(false, (A), false, (B), (short)0, (C), false, false)

// ---------------------------------------------------------------------------
// Gate: scores = x @ gate_w^T ; softmax ; +bias ; top-6 ; dense combine-weights
// Also converts x (f32) -> xb (bf16). One wave per token.
// ---------------------------------------------------------------------------
__global__ void __launch_bounds__(256)
gate_kernel(const float* __restrict__ x, const float* __restrict__ gate_w,
            const float* __restrict__ gate_b, float* __restrict__ cw,
            u16* __restrict__ xb)
{
    const int lane = threadIdx.x & 31;
    const int wv   = threadIdx.x >> 5;
    const int t    = blockIdx.x * 8 + wv;

    const float* xr = x + (size_t)t * DIM;
    float acc[NEXP];
#pragma unroll
    for (int e = 0; e < NEXP; ++e) acc[e] = 0.f;

    for (int d = lane * 4; d < DIM; d += 128) {
        const float4 xv = *(const float4*)(xr + d);
        uint2 packed;
        packed.x = pk2bf(xv.x, xv.y);
        packed.y = pk2bf(xv.z, xv.w);
        *(uint2*)(xb + (size_t)t * DIM + d) = packed;
#pragma unroll
        for (int e = 0; e < NEXP; ++e) {
            const float4 gv = *(const float4*)(gate_w + (size_t)e * DIM + d);
            acc[e] += xv.x * gv.x + xv.y * gv.y + xv.z * gv.z + xv.w * gv.w;
        }
    }
#pragma unroll
    for (int e = 0; e < NEXP; ++e) {
#pragma unroll
        for (int off = 16; off > 0; off >>= 1)
            acc[e] += __shfl_xor(acc[e], off, 32);
    }
    if (lane == 0) {
        float mx = acc[0];
#pragma unroll
        for (int e = 1; e < NEXP; ++e) mx = fmaxf(mx, acc[e]);
        float prob[NEXP], biased[NEXP], sum = 0.f;
#pragma unroll
        for (int e = 0; e < NEXP; ++e) { prob[e] = __expf(acc[e] - mx); sum += prob[e]; }
        const float inv = 1.f / sum;
        float row[NEXP];
#pragma unroll
        for (int e = 0; e < NEXP; ++e) {
            prob[e] *= inv;
            biased[e] = prob[e] + gate_b[e];
            row[e] = 0.f;
        }
        for (int k = 0; k < TOPK; ++k) {
            int bj = 0; float bv = -3.4e38f;
#pragma unroll
            for (int e = 0; e < NEXP; ++e)
                if (biased[e] > bv) { bv = biased[e]; bj = e; }
            row[bj] = prob[bj];           // ROUTE_SCALE = 1
            biased[bj] = -3.4e38f;
        }
#pragma unroll
        for (int e = 0; e < NEXP; ++e) cw[(size_t)t * NEXP + e] = row[e];
    }
}

// ---------------------------------------------------------------------------
// h = silu(xb @ W1^T + B1) * (xb @ W3^T + B3)   ->  hbuf (bf16, [T, N])
// Block tile 128x128 (waves 4M x 2N, each 32x64); A/B staged in LDS as bf16,
// weights converted f32->bf16 exactly once per block.
// ---------------------------------------------------------------------------
__global__ void __launch_bounds__(256)
moe_h_kernel(const u16* __restrict__ xb,
             const float* __restrict__ W1, const float* __restrict__ B1,
             const float* __restrict__ W3, const float* __restrict__ B3,
             u16* __restrict__ hbuf, int N, int K)
{
    __shared__ u16 smA [BM * LDS_ROW];
    __shared__ u16 smB1[BN * LDS_ROW];
    __shared__ u16 smB3[BN * LDS_ROW];

    const int tid  = threadIdx.x;
    const int lane = tid & 31;
    const int wv   = tid >> 5;
    const int hlf  = lane >> 4;
    const int r    = lane & 15;
    const int wvM  = wv & 3;          // 0..3  -> M sub-tile
    const int wvN  = wv >> 2;         // 0..1  -> N sub-tile
    const int blkM = blockIdx.x * BM;
    const int blkN = blockIdx.y * BN;

    // cooperative loader mapping: 2 threads per row, 32 elements each
    const int lrow = tid >> 1;            // 0..127
    const int lcol = (tid & 1) * 32;      // 0 | 32

    const u16*   gA  = xb + (size_t)(blkM + lrow) * K + lcol;
    const float* gB1 = W1 + (size_t)(blkN + lrow) * K + lcol;
    const float* gB3 = W3 + (size_t)(blkN + lrow) * K + lcol;

    u16* sA  = &smA [lrow * LDS_ROW + lcol];
    u16* sB1 = &smB1[lrow * LDS_ROW + lcol];
    u16* sB3 = &smB3[lrow * LDS_ROW + lcol];

    v8f acc1[2][4], acc3[2][4];
    const v8f vzero = {0.f, 0.f, 0.f, 0.f, 0.f, 0.f, 0.f, 0.f};
#pragma unroll
    for (int j = 0; j < 4; ++j) {
        acc1[0][j] = vzero; acc1[1][j] = vzero;
        acc3[0][j] = vzero; acc3[1][j] = vzero;
    }

    for (int kb = 0; kb < K; kb += BK) {
        __syncthreads();
        stage_row_bf16(sA,  gA  + kb);
        stage_row_f32 (sB1, gB1 + kb);
        stage_row_f32 (sB3, gB3 + kb);
        __syncthreads();

#pragma unroll
        for (int kc = 0; kc < 2; ++kc) {
            const int ko = kc * 32 + hlf * 8;
            FragBF a0, a1;
            load_frag_lds(a0, &smA[(wvM * 32      + r) * LDS_ROW + ko]);
            load_frag_lds(a1, &smA[(wvM * 32 + 16 + r) * LDS_ROW + ko]);
#pragma unroll
            for (int j = 0; j < 4; ++j) {
                const int brow = (wvN * 64 + j * 16 + r) * LDS_ROW + ko;
                FragBF b1, b3;
                load_frag_lds(b1, &smB1[brow]);
                acc1[0][j] = WMMA_BF16(a0.bf, b1.bf, acc1[0][j]);
                acc1[1][j] = WMMA_BF16(a1.bf, b1.bf, acc1[1][j]);
                load_frag_lds(b3, &smB3[brow]);
                acc3[0][j] = WMMA_BF16(a0.bf, b3.bf, acc3[0][j]);
                acc3[1][j] = WMMA_BF16(a1.bf, b3.bf, acc3[1][j]);
            }
        }
    }

    // epilogue: silu(a1)*a3, write bf16
    const int tileM = blkM + wvM * 32;
    const int tileN = blkN + wvN * 64;
#pragma unroll
    for (int mi = 0; mi < 2; ++mi) {
#pragma unroll
        for (int j = 0; j < 4; ++j) {
            const int n = tileN + j * 16 + r;
            const float b1v = B1[n];
            const float b3v = B3[n];
#pragma unroll
            for (int v = 0; v < 8; ++v) {
                const int m = tileM + mi * 16 + v + 8 * hlf;
                const float ax = acc1[mi][j][v] + b1v;
                const float g  = ax * (1.f / (1.f + __expf(-ax)));  // silu
                const float hv = g * (acc3[mi][j][v] + b3v);
                hbuf[(size_t)m * N + n] = f2bf(hv);
            }
        }
    }
}

// ---------------------------------------------------------------------------
// out (+)= cw[:,e] * (hbuf @ W2^T + B2)
// accumulate==0: plain store (shared expert initializes d_out).
// ---------------------------------------------------------------------------
__global__ void __launch_bounds__(256)
moe_out_kernel(const u16* __restrict__ hbuf,
               const float* __restrict__ W2, const float* __restrict__ B2,
               const float* __restrict__ cw, int e,
               float* __restrict__ out, int N, int K, int accumulate)
{
    __shared__ u16 smA[BM * LDS_ROW];
    __shared__ u16 smB[BN * LDS_ROW];

    const int tid  = threadIdx.x;
    const int lane = tid & 31;
    const int wv   = tid >> 5;
    const int hlf  = lane >> 4;
    const int r    = lane & 15;
    const int wvM  = wv & 3;
    const int wvN  = wv >> 2;
    const int blkM = blockIdx.x * BM;
    const int blkN = blockIdx.y * BN;

    const int lrow = tid >> 1;
    const int lcol = (tid & 1) * 32;

    const u16*   gA = hbuf + (size_t)(blkM + lrow) * K + lcol;
    const float* gB = W2   + (size_t)(blkN + lrow) * K + lcol;

    u16* sA = &smA[lrow * LDS_ROW + lcol];
    u16* sB = &smB[lrow * LDS_ROW + lcol];

    v8f acc[2][4];
    const v8f vzero = {0.f, 0.f, 0.f, 0.f, 0.f, 0.f, 0.f, 0.f};
#pragma unroll
    for (int j = 0; j < 4; ++j) { acc[0][j] = vzero; acc[1][j] = vzero; }

    for (int kb = 0; kb < K; kb += BK) {
        __syncthreads();
        stage_row_bf16(sA, gA + kb);
        stage_row_f32 (sB, gB + kb);
        __syncthreads();

#pragma unroll
        for (int kc = 0; kc < 2; ++kc) {
            const int ko = kc * 32 + hlf * 8;
            FragBF a0, a1;
            load_frag_lds(a0, &smA[(wvM * 32      + r) * LDS_ROW + ko]);
            load_frag_lds(a1, &smA[(wvM * 32 + 16 + r) * LDS_ROW + ko]);
#pragma unroll
            for (int j = 0; j < 4; ++j) {
                FragBF b;
                load_frag_lds(b, &smB[(wvN * 64 + j * 16 + r) * LDS_ROW + ko]);
                acc[0][j] = WMMA_BF16(a0.bf, b.bf, acc[0][j]);
                acc[1][j] = WMMA_BF16(a1.bf, b.bf, acc[1][j]);
            }
        }
    }

    const int tileM = blkM + wvM * 32;
    const int tileN = blkN + wvN * 64;
#pragma unroll
    for (int mi = 0; mi < 2; ++mi) {
#pragma unroll
        for (int j = 0; j < 4; ++j) {
            const int n = tileN + j * 16 + r;
            const float b2v = B2[n];
#pragma unroll
            for (int v = 0; v < 8; ++v) {
                const int m = tileM + mi * 16 + v + 8 * hlf;
                float val = acc[mi][j][v] + b2v;
                if (cw) val *= cw[(size_t)m * NEXP + e];
                const size_t o = (size_t)m * N + n;
                if (accumulate) out[o] += val;
                else            out[o]  = val;
            }
        }
    }
}

// ---------------------------------------------------------------------------
extern "C" void kernel_launch(void* const* d_in, const int* in_sizes, int n_in,
                              void* d_out, int out_size, void* d_ws, size_t ws_size,
                              hipStream_t stream)
{
    const float* x      = (const float*)d_in[0];
    const float* gate_w = (const float*)d_in[1];
    const float* gate_b = (const float*)d_in[2];
    const float* w1     = (const float*)d_in[3];
    const float* b1     = (const float*)d_in[4];
    const float* w2     = (const float*)d_in[5];
    const float* b2     = (const float*)d_in[6];
    const float* w3     = (const float*)d_in[7];
    const float* b3     = (const float*)d_in[8];
    const float* ws1    = (const float*)d_in[9];
    const float* bs1    = (const float*)d_in[10];
    const float* ws2    = (const float*)d_in[11];
    const float* bs2    = (const float*)d_in[12];
    const float* ws3    = (const float*)d_in[13];
    const float* bs3    = (const float*)d_in[14];
    float* out = (float*)d_out;

    // workspace layout
    char* ws = (char*)d_ws;
    const size_t CW_BYTES = (size_t)T_TOK * NEXP * sizeof(float);   // 256 KB
    const size_t XB_BYTES = (size_t)T_TOK * DIM * sizeof(u16);      // 16 MB
    float* cw   = (float*)ws;
    u16*   xb   = (u16*)(ws + CW_BYTES);
    u16*   hbuf = (u16*)(ws + CW_BYTES + XB_BYTES);                 // T x HS bf16 (23 MB)

    // 1) gate + x->bf16 conversion
    gate_kernel<<<T_TOK / 8, 256, 0, stream>>>(x, gate_w, gate_b, cw, xb);

    // 2) shared expert: h_s then out = h_s @ ws2^T + bs2 (store-mode, inits d_out)
    moe_h_kernel<<<dim3(T_TOK / BM, HSH / BN), 256, 0, stream>>>(
        xb, ws1, bs1, ws3, bs3, hbuf, HSH, DIM);
    moe_out_kernel<<<dim3(T_TOK / BM, DIM / BN), 256, 0, stream>>>(
        hbuf, ws2, bs2, nullptr, 0, out, DIM, HSH, /*accumulate=*/0);

    // 3) routed experts: dense scan (matches reference; cw rows are 0 for
    //    non-top-k experts), sequential on the stream so hbuf is reused.
    for (int e = 0; e < NEXP; ++e) {
        const float* W1e = w1 + (size_t)e * HID * DIM;
        const float* B1e = b1 + (size_t)e * HID;
        const float* W3e = w3 + (size_t)e * HID * DIM;
        const float* B3e = b3 + (size_t)e * HID;
        const float* W2e = w2 + (size_t)e * DIM * HID;
        const float* B2e = b2 + (size_t)e * DIM;
        moe_h_kernel<<<dim3(T_TOK / BM, HID / BN), 256, 0, stream>>>(
            xb, W1e, B1e, W3e, B3e, hbuf, HID, DIM);
        moe_out_kernel<<<dim3(T_TOK / BM, DIM / BN), 256, 0, stream>>>(
            hbuf, W2e, B2e, cw, e, out, DIM, HID, /*accumulate=*/1);
    }
}
